// Int8LayerNorm_19765439496651
// MI455X (gfx1250) — 8-bit matrix operand order repaired and checked by reading, compiled, not run on hardware
//
#include <hip/hip_runtime.h>
#include <hip/hip_bf16.h>


typedef __attribute__((ext_vector_type(8))) int v8i;

#define N_FEAT   4096
#define N_ROWS   8192
#define ROWS_PER_WAVE 16
#define WAVES_PER_BLK 8
#define TILE_K   64

__device__ __forceinline__ float quant_clip(float x, float inv_s) {
    float q = rintf(x * inv_s);
    return fminf(fmaxf(q, -127.0f), 127.0f);
}

__device__ __forceinline__ float eval_y(float x, float s, float inv_s,
                                        float mu, float inv_std,
                                        float g, float b) {
    float xq = quant_clip(x, inv_s) * s;
    return (xq - mu) * inv_std * g + b;
}

__device__ __forceinline__ float scale_from_bits(unsigned bits) {
    return fmaxf(__uint_as_float(bits) / 127.0f, 1e-8f);
}

__global__ void k0_init(unsigned* wsu) {
    if (threadIdx.x == 0) { wsu[0] = 0u; wsu[1] = 0u; }
}

__global__ void k1_amax_x(const float* __restrict__ x, unsigned* __restrict__ wsu,
                          int n4) {
    __shared__ float smax[WAVES_PER_BLK];
    const float4* x4 = (const float4*)x;
    float m = 0.0f;
    for (int i = blockIdx.x * blockDim.x + threadIdx.x; i < n4;
         i += gridDim.x * blockDim.x) {
        float4 v = x4[i];
        m = fmaxf(m, fmaxf(fmaxf(fabsf(v.x), fabsf(v.y)),
                           fmaxf(fabsf(v.z), fabsf(v.w))));
    }
    #pragma unroll
    for (int off = 16; off > 0; off >>= 1)
        m = fmaxf(m, __shfl_xor(m, off, 32));
    int wid = threadIdx.x >> 5;
    if ((threadIdx.x & 31) == 0) smax[wid] = m;
    __syncthreads();
    if (threadIdx.x == 0) {
        float bm = 0.0f;
        #pragma unroll
        for (int w = 0; w < WAVES_PER_BLK; ++w) bm = fmaxf(bm, smax[w]);
        atomicMax(&wsu[0], __float_as_uint(bm));
    }
}

__global__ void __launch_bounds__(256)
k2_row_stats(const float* __restrict__ x,
             const float* __restrict__ gamma,
             const float* __restrict__ beta,
             unsigned* __restrict__ wsu,
             float2* __restrict__ rowstats) {
    __shared__ int  tileLds[WAVES_PER_BLK][ROWS_PER_WAVE][TILE_K / 4];
    __shared__ int  csumLds[WAVES_PER_BLK][16][16];
    __shared__ int  csqLds [WAVES_PER_BLK][16][16];
    __shared__ float muLds [WAVES_PER_BLK][16];
    __shared__ float invLds[WAVES_PER_BLK][16];

    const int lane = threadIdx.x & 31;
    const int wid  = threadIdx.x >> 5;
    const int gw   = blockIdx.x * WAVES_PER_BLK + wid;
    const int r0   = gw * ROWS_PER_WAVE;

    const float s     = scale_from_bits(wsu[0]);
    const float inv_s = 1.0f / s;

    const int myrow = lane & 15;
    const int h     = lane >> 4;

    v8i ones;
    #pragma unroll
    for (int i = 0; i < 8; ++i) ones[i] = 0x01010101;
    v8i accS, accQ;
    #pragma unroll
    for (int i = 0; i < 8; ++i) { accS[i] = 0; accQ[i] = 0; }

    for (int t = 0; t < N_FEAT / TILE_K; ++t) {
        const int c0 = t * TILE_K;
        const float4* src =
            (const float4*)(x + (size_t)(r0 + myrow) * N_FEAT + c0 + h * 32);
        #pragma unroll
        for (int j = 0; j < 8; ++j) {
            float4 v = src[j];
            int q0 = (int)quant_clip(v.x, inv_s);
            int q1 = (int)quant_clip(v.y, inv_s);
            int q2 = (int)quant_clip(v.z, inv_s);
            int q3 = (int)quant_clip(v.w, inv_s);
            tileLds[wid][myrow][h * 8 + j] =
                (q0 & 0xff) | ((q1 & 0xff) << 8) | ((q2 & 0xff) << 16)
                | ((q3 & 0xff) << 24);
        }
        __syncthreads();

        const int* rowp = &tileLds[wid][myrow][0];
        v8i A;
        #pragma unroll
        for (int g = 0; g < 4; ++g) {
            int2 p = *(const int2*)(rowp + h * 4 + (g & 1) * 2 + (g >> 1) * 8);
            A[2 * g]     = p.x;
            A[2 * g + 1] = p.y;
        }
        v8i B;
        {
            int4 q0 = *(const int4*)(rowp + h * 4);
            int4 q1 = *(const int4*)(rowp + 8 + h * 4);
            B[0] = q0.x; B[1] = q0.y; B[2] = q0.z; B[3] = q0.w;
            B[4] = q1.x; B[5] = q1.y; B[6] = q1.z; B[7] = q1.w;
        }
        accS = __builtin_amdgcn_wmma_i32_16x16x64_iu8(
            true, A, false, ones, accS, false, false);
        accQ = __builtin_amdgcn_wmma_i32_16x16x64_iu8(
            true, A, true, B, accQ, false, false);
        __syncthreads();
    }

    #pragma unroll
    for (int v = 0; v < 8; ++v) {
        csumLds[wid][8 * h + v][myrow] = accS[v];
        csqLds [wid][8 * h + v][myrow] = accQ[v];
    }
    __syncthreads();

    if (lane < 16) {
        int ex_i  = csumLds[wid][lane][0];
        int ex2_i = csqLds [wid][lane][lane];
        float mu  = ((float)ex_i * s) / (float)N_FEAT;
        float ex2 = (float)ex2_i * s * s;
        float var = fmaxf(ex2 / (float)N_FEAT - mu * mu, 0.0f);
        float vr  = fminf(fmaxf(rintf(var), 1.0f), 65535.0f);
        int d = (int)vr;
        int q = (int)floorf(sqrtf((float)d));
        if ((q + 1) * (q + 1) <= d) ++q;
        if (q * q > d) --q;
        int r = (d > q * q + q) ? (q + 1) : q;
        float inv_std = 1.0f / (float)r;
        muLds [wid][lane] = mu;
        invLds[wid][lane] = inv_std;
        rowstats[r0 + lane] = make_float2(mu, inv_std);
    }
    __syncthreads();

    float mx = 0.0f;
    const int n4 = ROWS_PER_WAVE * N_FEAT / 4;
    for (int i = lane; i < n4; i += 32) {
        int row = i >> 10;
        int col = (i & 1023) * 4;
        float4 xv = *(const float4*)(x + (size_t)(r0 + row) * N_FEAT + col);
        float4 gv = *(const float4*)(gamma + col);
        float4 bv = *(const float4*)(beta + col);
        float mu = muLds[wid][row], iv = invLds[wid][row];
        mx = fmaxf(mx, fabsf(eval_y(xv.x, s, inv_s, mu, iv, gv.x, bv.x)));
        mx = fmaxf(mx, fabsf(eval_y(xv.y, s, inv_s, mu, iv, gv.y, bv.y)));
        mx = fmaxf(mx, fabsf(eval_y(xv.z, s, inv_s, mu, iv, gv.z, bv.z)));
        mx = fmaxf(mx, fabsf(eval_y(xv.w, s, inv_s, mu, iv, gv.w, bv.w)));
    }
    #pragma unroll
    for (int off = 16; off > 0; off >>= 1)
        mx = fmaxf(mx, __shfl_xor(mx, off, 32));
    if (lane == 0) atomicMax(&wsu[1], __float_as_uint(mx));
}

__global__ void __launch_bounds__(256)
k3_output(const float* __restrict__ x,
          const float* __restrict__ gamma,
          const float* __restrict__ beta,
          const unsigned* __restrict__ wsu,
          const float2* __restrict__ rowstats,
          float* __restrict__ out) {
    const float s      = scale_from_bits(wsu[0]);
    const float inv_s  = 1.0f / s;
    const float so     = scale_from_bits(wsu[1]);
    const float inv_so = 1.0f / so;

    int i = blockIdx.x * blockDim.x + threadIdx.x;
    int row = i >> 10;
    int col = (i & 1023) * 4;
    float4 xv = *(const float4*)(x + (size_t)row * N_FEAT + col);
    float4 gv = *(const float4*)(gamma + col);
    float4 bv = *(const float4*)(beta + col);
    float2 rs = rowstats[row];
    float4 o;
    o.x = quant_clip(eval_y(xv.x, s, inv_s, rs.x, rs.y, gv.x, bv.x), inv_so) * so;
    o.y = quant_clip(eval_y(xv.y, s, inv_s, rs.x, rs.y, gv.y, bv.y), inv_so) * so;
    o.z = quant_clip(eval_y(xv.z, s, inv_s, rs.x, rs.y, gv.z, bv.z), inv_so) * so;
    o.w = quant_clip(eval_y(xv.w, s, inv_s, rs.x, rs.y, gv.w, bv.w), inv_so) * so;
    *(float4*)(out + (size_t)row * N_FEAT + col) = o;
}

extern "C" void kernel_launch(void* const* d_in, const int* in_sizes, int n_in,
                              void* d_out, int out_size, void* d_ws, size_t ws_size,
                              hipStream_t stream) {
    const float* x     = (const float*)d_in[0];
    const float* gamma = (const float*)d_in[1];
    const float* beta  = (const float*)d_in[2];
    float* out = (float*)d_out;

    unsigned* wsu      = (unsigned*)d_ws;
    float2*   rowstats = (float2*)((char*)d_ws + 16);

    const int n_elem = N_ROWS * N_FEAT;
    const int n4     = n_elem / 4;

    k0_init<<<1, 32, 0, stream>>>(wsu);
    k1_amax_x<<<2048, 256, 0, stream>>>(x, wsu, n4);
    k2_row_stats<<<N_ROWS / (ROWS_PER_WAVE * WAVES_PER_BLK), 256, 0, stream>>>(
        x, gamma, beta, wsu, rowstats);
    k3_output<<<n4 / 256, 256, 0, stream>>>(x, gamma, beta, wsu, rowstats, out);
}
